// MQANDecoder_90074054132383
// MI455X (gfx1250) — matrix operand order repaired and checked by reading, not compiled, not run on hardware
//
#include <hip/hip_runtime.h>
#include <math.h>

// ---------------------------------------------------------------------------
// MQAN decoder (LSTM + dot attention, input feeding) for MI455X / gfx1250.
// B=64, T=64, D=512, SC=512, SQ=128.
// GEMMs vs shared weights use v_wmma_f32_16x16x32_bf16 (wave32 WMMA),
// one wave per 16x64 output tile (4 accumulators share one A fragment).
// Per-batch attention GEMVs use coalesced f32 VALU dot products (L2 resident).
// ---------------------------------------------------------------------------

typedef __attribute__((ext_vector_type(16))) __bf16 v16bf;
typedef __attribute__((ext_vector_type(8)))  __bf16 v8bf;
typedef __attribute__((ext_vector_type(8)))  float  v8f;

#define B_  64
#define T_  64
#define D_  512
#define SC_ 512
#define SQ_ 128
#define G4D 2048   // 4*D
#define KALL 1536  // 3*D  (x_t | ctx_out | h)

__device__ __forceinline__ float sigmoidf_(float x) { return 1.0f / (1.0f + expf(-x)); }

// ---------------------------------------------------------------------------
// bf16 WMMA GEMM:  C[M,N] = act( A[M,K] * W[N,K]^T + bias )
// One wave computes a 16x64 tile: 4 N-accumulators reuse a single A fragment
// per 32-wide K step (2 A loads + 8 B loads per 4 WMMAs).
// K must be a multiple of 32; N a multiple of 64; M a multiple of 16.
// Optionally mirrors the result to a second destination C2 (row stride ldc2).
// ---------------------------------------------------------------------------
__global__ __launch_bounds__(32) void wmma_gemm_bf16(
    const __bf16* __restrict__ A, int lda,
    const __bf16* __restrict__ W, int ldw,
    const float*  __restrict__ bias,
    float* __restrict__ C,  int ldc,
    float* __restrict__ C2, int ldc2,
    int K, int act)
{
  const int lane = threadIdx.x & 31;
  const int m0 = blockIdx.y << 4;
  const int n0 = blockIdx.x << 6;           // 64 columns per wave

  // A fragment addressing (lane = row, 16 contig K as for B; not the ISA table order)
  const int arow = m0 + (lane & 15);
  const int akb  = (lane >> 4) << 4;        // 0 or 16
  // B fragment addressing (16-bit B 32x16 layout): lane = column, 16 contig K
  const int bcol = n0 + (lane & 15);
  const int bkb  = (lane >> 4) << 4;        // 0 or 16

  const __bf16* Ap = A + (long)arow * lda + akb;
  const __bf16* Wp = W + (long)bcol * ldw + bkb;
  const long wstep = (long)16 * ldw;        // next 16-column group

  v8f acc[4] = {};
  for (int k0 = 0; k0 < K; k0 += 32) {
    __builtin_prefetch(Wp + k0 + 64, 0, 0);               // global_prefetch_b8
    v16bf a = *(const v16bf*)(Ap + k0);                   // K+0..15 (or +16..31)
#pragma unroll
    for (int j = 0; j < 4; ++j) {
      v16bf b = *(const v16bf*)(Wp + (long)j * wstep + k0);
      acc[j] = __builtin_amdgcn_wmma_f32_16x16x32_bf16(
                   /*neg_a=*/false, a, /*neg_b=*/false, b,
                   /*c_mod=*/(short)0, acc[j], /*reuse_a=*/false, /*reuse_b=*/false);
    }
  }

  // C/D f32 16x16 layout: VGPR r -> M = r (+8 for upper lane half), lane&15 -> N
  const int rowbase = m0 + ((lane >> 4) << 3);
#pragma unroll
  for (int j = 0; j < 4; ++j) {
    const int col = n0 + (j << 4) + (lane & 15);
    const float bv = bias ? bias[col] : 0.0f;
#pragma unroll
    for (int r = 0; r < 8; ++r) {
      float v = acc[j][r] + bv;
      if (act) v = tanhf(v);
      C[(long)(rowbase + r) * ldc + col] = v;
      if (C2) C2[(long)(rowbase + r) * ldc2 + col] = v;
    }
  }
}

// ---------------------------------------------------------------------------
// Per-step element-wise / reduction kernels
// ---------------------------------------------------------------------------
__global__ void packA_kernel(const float* __restrict__ inp,
                             const float* __restrict__ ctx_out,
                             const float* __restrict__ h,
                             __bf16* __restrict__ Acat, int t)
{
  int idx = blockIdx.x * blockDim.x + threadIdx.x;
  if (idx >= B_ * KALL) return;
  int b = idx / KALL, k = idx % KALL;
  float v;
  if (k < D_)            v = inp[((long)b * T_ + t) * D_ + k];
  else if (k < 2 * D_)   v = ctx_out[b * D_ + (k - D_)];
  else                   v = h[b * D_ + (k - 2 * D_)];
  Acat[idx] = (__bf16)v;
}

__global__ void lstm_kernel(const float* __restrict__ gates,
                            float* __restrict__ h, float* __restrict__ c)
{
  int idx = blockIdx.x * blockDim.x + threadIdx.x;
  if (idx >= B_ * D_) return;
  int b = idx >> 9, j = idx & (D_ - 1);
  const float* g = gates + (long)b * G4D;
  float iv = sigmoidf_(g[j]);
  float fv = sigmoidf_(g[j + D_]);
  float gv = tanhf(g[j + 2 * D_]);
  float ov = sigmoidf_(g[j + 3 * D_]);
  float cn = fv * c[idx] + iv * gv;
  c[idx] = cn;
  h[idx] = ov * tanhf(cn);
}

__global__ void scores_kernel(const float* __restrict__ h,
                              const float* __restrict__ mem,
                              const unsigned char* __restrict__ mask,
                              float* __restrict__ scores, int S)
{
  int idx = blockIdx.x * blockDim.x + threadIdx.x;
  if (idx >= B_ * S) return;
  int b = idx / S, s = idx % S;
  const float4* hp = (const float4*)(h + (long)b * D_);
  const float4* mp = (const float4*)(mem + ((long)b * S + s) * D_);
  float acc = 0.0f;
#pragma unroll 4
  for (int i = 0; i < D_ / 4; ++i) {
    float4 a = hp[i], m4 = mp[i];
    acc += a.x * m4.x + a.y * m4.y + a.z * m4.z + a.w * m4.w;
  }
  scores[idx] = mask[idx] ? -__builtin_inff() : acc;
}

__global__ void softmax_kernel(const float* __restrict__ scores,
                               float* __restrict__ attn,
                               float* __restrict__ out_attn,  // d_out attn slab
                               int S, int t)
{
  int b = blockIdx.x;
  int tid = threadIdx.x;  // 256 threads
  __shared__ float red[256];
  const float* sr = scores + (long)b * S;

  float m = -__builtin_inff();
  for (int s = tid; s < S; s += 256) m = fmaxf(m, sr[s]);
  red[tid] = m; __syncthreads();
  for (int off = 128; off > 0; off >>= 1) {
    if (tid < off) red[tid] = fmaxf(red[tid], red[tid + off]);
    __syncthreads();
  }
  m = red[0]; __syncthreads();

  float sum = 0.0f;
  for (int s = tid; s < S; s += 256) sum += expf(sr[s] - m);
  red[tid] = sum; __syncthreads();
  for (int off = 128; off > 0; off >>= 1) {
    if (tid < off) red[tid] += red[tid + off];
    __syncthreads();
  }
  float inv = 1.0f / red[0];

  float* ab = attn + (long)b * S;
  float* ob = out_attn + ((long)b * T_ + t) * S;
  for (int s = tid; s < S; s += 256) {
    float v = expf(sr[s] - m) * inv;
    ab[s] = v; ob[s] = v;
  }
}

__global__ void align_kernel(const float* __restrict__ attn,
                             const float* __restrict__ mem,
                             float* __restrict__ al, int S)
{
  int idx = blockIdx.x * blockDim.x + threadIdx.x;
  if (idx >= B_ * D_) return;
  int b = idx >> 9, d = idx & (D_ - 1);
  const float* ab = attn + (long)b * S;
  const float* mb = mem + (long)b * S * D_ + d;
  float acc = 0.0f;
  for (int s = 0; s < S; ++s) acc += ab[s] * mb[(long)s * D_];
  al[idx] = acc;
}

__global__ void packHA_kernel(const float* __restrict__ h,
                              const float* __restrict__ al,
                              __bf16* __restrict__ catbuf)
{
  int idx = blockIdx.x * blockDim.x + threadIdx.x;
  if (idx >= B_ * 2 * D_) return;
  int b = idx >> 10, k = idx & (2 * D_ - 1);
  float v = (k < D_) ? h[b * D_ + k] : al[b * D_ + (k - D_)];
  catbuf[idx] = (__bf16)v;
}

__global__ void switch_kernel(const float* __restrict__ h,
                              const float* __restrict__ ctx_out,
                              const float* __restrict__ q_out,
                              const float* __restrict__ inp,
                              float* __restrict__ vps,
                              float* __restrict__ cqs, int t)
{
  int idx = blockIdx.x * blockDim.x + threadIdx.x;
  if (idx >= B_ * 3 * D_) return;
  int b = idx / (3 * D_), k = idx % (3 * D_);
  long o = ((long)b * T_ + t) * (3 * D_) + k;
  if (k < D_) {
    float hv = h[b * D_ + k];
    vps[o] = hv; cqs[o] = hv;
  } else if (k < 2 * D_) {
    int j = k - D_;
    vps[o] = ctx_out[b * D_ + j];
    cqs[o] = q_out[b * D_ + j];
  } else {
    float x = inp[((long)b * T_ + t) * D_ + (k - 2 * D_)];
    vps[o] = x; cqs[o] = x;
  }
}

// ---------------------------------------------------------------------------
// One-time (per launch) prep kernels
// ---------------------------------------------------------------------------
__global__ void prep_wall_kernel(const float* __restrict__ W_ih,   // [2048][1024]
                                 const float* __restrict__ W_hh,   // [2048][512]
                                 __bf16* __restrict__ Wall)        // [2048][1536]
{
  int idx = blockIdx.x * blockDim.x + threadIdx.x;
  if (idx >= G4D * KALL) return;
  int n = idx / KALL, k = idx % KALL;
  float v = (k < 2 * D_) ? W_ih[(long)n * (2 * D_) + k]
                         : W_hh[(long)n * D_ + (k - 2 * D_)];
  Wall[idx] = (__bf16)v;
}

__global__ void cvt_bf16_kernel(const float* __restrict__ src,
                                __bf16* __restrict__ dst, int n)
{
  int idx = blockIdx.x * blockDim.x + threadIdx.x;
  if (idx < n) dst[idx] = (__bf16)src[idx];
}

__global__ void bias_init_kernel(const float* __restrict__ b_ih,
                                 const float* __restrict__ b_hh,
                                 float* __restrict__ bsum)
{
  int idx = blockIdx.x * blockDim.x + threadIdx.x;
  if (idx < G4D) bsum[idx] = b_ih[idx] + b_hh[idx];
}

__global__ void state_init_kernel(const float* __restrict__ h0,
                                  const float* __restrict__ c0,
                                  float* __restrict__ h, float* __restrict__ c,
                                  float* __restrict__ ctx_out)
{
  int idx = blockIdx.x * blockDim.x + threadIdx.x;
  if (idx >= B_ * D_) return;
  h[idx] = h0[idx];
  c[idx] = c0[idx];
  ctx_out[idx] = 0.0f;  // make_init_output -> zeros
}

__global__ void final_kernel(const float* __restrict__ h, const float* __restrict__ c,
                             float* __restrict__ oh, float* __restrict__ oc)
{
  int idx = blockIdx.x * blockDim.x + threadIdx.x;
  if (idx >= B_ * D_) return;
  oh[idx] = h[idx];
  oc[idx] = c[idx];
}

// ---------------------------------------------------------------------------
// Host side
// ---------------------------------------------------------------------------
extern "C" void kernel_launch(void* const* d_in, const int* in_sizes, int n_in,
                              void* d_out, int out_size, void* d_ws, size_t ws_size,
                              hipStream_t stream)
{
  (void)in_sizes; (void)n_in; (void)out_size; (void)ws_size;

  const float* inp   = (const float*)d_in[0];   // [B,T,D]
  const float* ctx   = (const float*)d_in[1];   // [B,SC,D]
  const float* ques  = (const float*)d_in[2];   // [B,SQ,D]
  const float* W_ih  = (const float*)d_in[3];   // [4D,2D]
  const float* W_hh  = (const float*)d_in[4];   // [4D,D]
  const float* b_ih  = (const float*)d_in[5];
  const float* b_hh  = (const float*)d_in[6];
  const float* W_co  = (const float*)d_in[7];   // [D,2D]
  const float* W_qo  = (const float*)d_in[8];   // [D,2D]
  const float* h0    = (const float*)d_in[9];
  const float* c0    = (const float*)d_in[10];
  const unsigned char* cmask = (const unsigned char*)d_in[11];  // [B,SC] bool
  const unsigned char* qmask = (const unsigned char*)d_in[12];  // [B,SQ] bool

  float* out = (float*)d_out;
  // output slab offsets (floats), in reference return order
  const size_t OFF_CTXOUTS = 0;                               // [B,T,D]
  const size_t OFF_VPS     = (size_t)B_ * T_ * D_;            // [B,T,3D]
  const size_t OFF_CQS     = OFF_VPS + (size_t)B_ * T_ * 3 * D_;
  const size_t OFF_CATT    = OFF_CQS + (size_t)B_ * T_ * 3 * D_;  // [B,T,SC]
  const size_t OFF_QATT    = OFF_CATT + (size_t)B_ * T_ * SC_;    // [B,T,SQ]
  const size_t OFF_H       = OFF_QATT + (size_t)B_ * T_ * SQ_;
  const size_t OFF_C       = OFF_H + (size_t)B_ * D_;

  // workspace carve-up (256B aligned)
  char* ws = (char*)d_ws;
  size_t cur = 0;
  auto carve = [&](size_t bytes) -> char* {
    char* p = ws + cur;
    cur = (cur + bytes + 255) & ~(size_t)255;
    return p;
  };
  __bf16* Wall    = (__bf16*)carve((size_t)G4D * KALL * 2);   // [2048][1536] bf16
  __bf16* Wco     = (__bf16*)carve((size_t)D_ * 2 * D_ * 2);  // [512][1024] bf16
  __bf16* Wqo     = (__bf16*)carve((size_t)D_ * 2 * D_ * 2);
  float*  bsum    = (float*) carve((size_t)G4D * 4);
  float*  gates   = (float*) carve((size_t)B_ * G4D * 4);
  float*  h       = (float*) carve((size_t)B_ * D_ * 4);
  float*  c       = (float*) carve((size_t)B_ * D_ * 4);
  float*  ctx_out = (float*) carve((size_t)B_ * D_ * 4);
  float*  q_out   = (float*) carve((size_t)B_ * D_ * 4);
  float*  al      = (float*) carve((size_t)B_ * D_ * 4);
  float*  scores  = (float*) carve((size_t)B_ * SC_ * 4);
  float*  attn    = (float*) carve((size_t)B_ * SC_ * 4);
  __bf16* Acat    = (__bf16*)carve((size_t)B_ * KALL * 2);    // [64][1536] bf16
  __bf16* catbuf  = (__bf16*)carve((size_t)B_ * 2 * D_ * 2);  // [64][1024] bf16

  const int TPB = 256;
  auto nb = [&](int n) { return (n + TPB - 1) / TPB; };

  // ---- prep (every call: determinism) ----
  prep_wall_kernel<<<nb(G4D * KALL), TPB, 0, stream>>>(W_ih, W_hh, Wall);
  cvt_bf16_kernel<<<nb(D_ * 2 * D_), TPB, 0, stream>>>(W_co, Wco, D_ * 2 * D_);
  cvt_bf16_kernel<<<nb(D_ * 2 * D_), TPB, 0, stream>>>(W_qo, Wqo, D_ * 2 * D_);
  bias_init_kernel<<<nb(G4D), TPB, 0, stream>>>(b_ih, b_hh, bsum);
  state_init_kernel<<<nb(B_ * D_), TPB, 0, stream>>>(h0, c0, h, c, ctx_out);

  // ---- sequential decode over T steps ----
  for (int t = 0; t < T_; ++t) {
    // gates = [x_t | ctx_out_prev | h] @ Wall^T + (b_ih + b_hh)
    packA_kernel<<<nb(B_ * KALL), TPB, 0, stream>>>(inp, ctx_out, h, Acat, t);
    wmma_gemm_bf16<<<dim3(G4D / 64, B_ / 16), 32, 0, stream>>>(
        Acat, KALL, Wall, KALL, bsum, gates, G4D, (float*)nullptr, 0, KALL, 0);
    lstm_kernel<<<nb(B_ * D_), TPB, 0, stream>>>(gates, h, c);

    // context attention
    scores_kernel<<<nb(B_ * SC_), TPB, 0, stream>>>(h, ctx, cmask, scores, SC_);
    softmax_kernel<<<B_, TPB, 0, stream>>>(scores, attn, out + OFF_CATT, SC_, t);
    align_kernel<<<nb(B_ * D_), TPB, 0, stream>>>(attn, ctx, al, SC_);
    packHA_kernel<<<nb(B_ * 2 * D_), TPB, 0, stream>>>(h, al, catbuf);
    // ctx_out = tanh([h|align] @ W_ctx_out^T); also write ctx_outs[b,t,:]
    wmma_gemm_bf16<<<dim3(D_ / 64, B_ / 16), 32, 0, stream>>>(
        catbuf, 2 * D_, Wco, 2 * D_, (const float*)nullptr,
        ctx_out, D_, out + OFF_CTXOUTS + (size_t)t * D_, T_ * D_, 2 * D_, 1);

    // question attention
    scores_kernel<<<nb(B_ * SQ_), TPB, 0, stream>>>(h, ques, qmask, scores, SQ_);
    softmax_kernel<<<B_, TPB, 0, stream>>>(scores, attn, out + OFF_QATT, SQ_, t);
    align_kernel<<<nb(B_ * D_), TPB, 0, stream>>>(attn, ques, al, SQ_);
    packHA_kernel<<<nb(B_ * 2 * D_), TPB, 0, stream>>>(h, al, catbuf);
    wmma_gemm_bf16<<<dim3(D_ / 64, B_ / 16), 32, 0, stream>>>(
        catbuf, 2 * D_, Wqo, 2 * D_, (const float*)nullptr,
        q_out, D_, (float*)nullptr, 0, 2 * D_, 1);

    // vps = [h, ctx_out, x_t]; cqs = [h, q_out, x_t]
    switch_kernel<<<nb(B_ * 3 * D_), TPB, 0, stream>>>(
        h, ctx_out, q_out, inp, out + OFF_VPS, out + OFF_CQS, t);
  }

  final_kernel<<<nb(B_ * D_), TPB, 0, stream>>>(h, c, out + OFF_H, out + OFF_C);
}
